// BiRnnCrf_31679678775679
// MI455X (gfx1250) — compile-verified, matrix operand order checked by reading, not run on hardware
//
#include <hip/hip_runtime.h>

typedef __attribute__((ext_vector_type(16))) _Float16 v16h;
typedef __attribute__((ext_vector_type(8)))  float    v8f;

#define LSEQ 512
#define BATCH 64
#define HSZ 256
#define HID 512
#define NC 5           // C = NTAGS + 2
#define START_TAG 3
#define STOP_TAG 4
#define IMPOSSIBLE_V (-1e4f)

__device__ __forceinline__ v8f vsplat8(float x) {
  v8f v;
  for (int i = 0; i < 8; ++i) v[i] = x;
  return v;
}

__device__ __forceinline__ float sigf(float x) {
  return 1.0f / (1.0f + __expf(-x));
}

// ---------------------------------------------------------------------------
// fp32 -> fp16 weight conversion
// ---------------------------------------------------------------------------
__global__ void cvt_f16_kernel(const float* __restrict__ src,
                               _Float16* __restrict__ dst, int n) {
  int i = blockIdx.x * blockDim.x + threadIdx.x;
  if (i < n) dst[i] = (_Float16)src[i];
}

// ---------------------------------------------------------------------------
// Embedding gather: xs[B][L] -> x0[L*B][256] fp16  (row = t*64 + b)
// ---------------------------------------------------------------------------
__global__ void embed_gather_kernel(const int* __restrict__ xs,
                                    const float* __restrict__ emb,
                                    _Float16* __restrict__ x0) {
  long long i = (long long)blockIdx.x * blockDim.x + threadIdx.x;
  if (i >= (long long)LSEQ * BATCH * 256) return;
  int e = (int)(i % 256);
  int b = (int)((i / 256) % BATCH);
  int t = (int)(i / (256LL * BATCH));
  int tok = xs[b * LSEQ + t];
  x0[i] = (_Float16)emb[(long long)tok * 256 + e];
}

// ---------------------------------------------------------------------------
// Input projection (fully parallel WMMA GEMM, 2x2 register blocking):
//   pre[L*B][2048] = x[L*B][Kin] @ Wih^T (both dirs) + bias
// 8 wave32s per block; each wave computes a 32x32 macro-tile (4 WMMA tiles),
// so each A/B fragment load is reused twice: 4 b128-load pairs -> 4 WMMAs.
// M = 32768 rows (t*64+b), N = 2048 (dir*1024 + gate*256 + hidden).
// ---------------------------------------------------------------------------
__global__ void __launch_bounds__(256)
inproj_kernel(const _Float16* __restrict__ x, int Kin,
              const _Float16* __restrict__ Wih,   // [2][1024][Kin] = [2048][Kin]
              const float* __restrict__ bias,     // [2][1024]
              _Float16* __restrict__ pre) {       // [L*B][2048]
  const int tid  = threadIdx.x;
  const int wave = tid >> 5;
  const int lane = tid & 31;
  const int l16  = lane & 15;
  const int hi   = lane >> 4;

  const long long mac = (long long)blockIdx.x * 8 + wave;
  const int NMT = 2048 / 32;             // 64 macro col tiles
  const int mm = (int)(mac / NMT);       // 0..1023 macro row tiles (32 rows)
  const int mn = (int)(mac % NMT);       // 0..63   macro col tiles (32 cols)

  const int row0 = mm * 32 + l16;        // M rows this lane supplies
  const int n0   = mn * 32 + l16;        // N cols this lane supplies
  const _Float16* xr0 = x + (size_t)row0 * Kin;
  const _Float16* xr1 = xr0 + (size_t)16 * Kin;
  const _Float16* wr0 = Wih + (size_t)n0 * Kin;
  const _Float16* wr1 = wr0 + (size_t)16 * Kin;

  v8f acc00 = vsplat8(0.0f), acc01 = vsplat8(0.0f);
  v8f acc10 = vsplat8(0.0f), acc11 = vsplat8(0.0f);

  const int kx = Kin >> 5;
  for (int kb = 0; kb < kx; ++kb) {
    const int ko = kb * 32 + hi * 16;
    v16h a0 = *(const v16h*)(xr0 + ko);
    v16h a1 = *(const v16h*)(xr1 + ko);
    v16h b0 = *(const v16h*)(wr0 + ko);
    v16h b1 = *(const v16h*)(wr1 + ko);
    acc00 = __builtin_amdgcn_wmma_f32_16x16x32_f16(false, a0, false, b0, (short)0, acc00, false, false);
    acc01 = __builtin_amdgcn_wmma_f32_16x16x32_f16(false, a0, false, b1, (short)0, acc01, false, false);
    acc10 = __builtin_amdgcn_wmma_f32_16x16x32_f16(false, a1, false, b0, (short)0, acc10, false, false);
    acc11 = __builtin_amdgcn_wmma_f32_16x16x32_f16(false, a1, false, b1, (short)0, acc11, false, false);
  }

  const float bv0 = bias[n0];
  const float bv1 = bias[n0 + 16];
  const int cbase = mn * 32 + l16;
#pragma unroll
  for (int r = 0; r < 8; ++r) {
    const int ra = mm * 32 + r + (hi << 3);       // WMMA C/D layout, tiles 0x
    const int rb = ra + 16;                       // tiles 1x
    pre[(size_t)ra * 2048 + cbase]      = (_Float16)(acc00[r] + bv0);
    pre[(size_t)ra * 2048 + cbase + 16] = (_Float16)(acc01[r] + bv1);
    pre[(size_t)rb * 2048 + cbase]      = (_Float16)(acc10[r] + bv0);
    pre[(size_t)rb * 2048 + cbase + 16] = (_Float16)(acc11[r] + bv1);
  }
}

// ---------------------------------------------------------------------------
// Sequential recurrent kernel (one direction per block, persistent over L).
//   pre  : [L*B][2048] fp16 precomputed x@Wih^T + b (dir*1024 + gate*256 + h)
//   Whh  : [2][1024][256] fp16
//   hout : [L][B][512] fp16 (dir d writes columns d*256..d*256+255)
// 32 wave32s; wave w: row tile rt=w&3, hidden tiles ct0=(w>>2)*2 .. +1.
// c-state in VGPRs, h-state in LDS across all 512 steps.
// ---------------------------------------------------------------------------
__global__ void __launch_bounds__(1024)
lstm_rec_kernel(const _Float16* __restrict__ pre,
                const _Float16* __restrict__ Whh,
                _Float16* __restrict__ hout) {
  const int dir  = blockIdx.x;
  const int tid  = threadIdx.x;
  const int wave = tid >> 5;
  const int lane = tid & 31;
  const int l16  = lane & 15;
  const int hi   = lane >> 4;
  const int rt   = wave & 3;             // batch-row tile
  const int ct0  = (wave >> 2) << 1;     // first of 2 owned hidden tiles

  __shared__ alignas(32) _Float16 hlds[64][272];  // 272 keeps rows 32B aligned
  for (int i = tid; i < 64 * 272; i += 1024)
    (&hlds[0][0])[i] = (_Float16)0.0f;
  __syncthreads();

  const _Float16* Wh = Whh + (size_t)dir * 1024 * 256;

  float creg[2][8];
  for (int s = 0; s < 2; ++s)
    for (int r = 0; r < 8; ++r) creg[s][r] = 0.0f;

  const int rowA = rt * 16 + l16;

  for (int t = 0; t < LSEQ; ++t) {
    const int tm = (dir == 1) ? (LSEQ - 1 - t) : t;

    // launder the per-lane column index so weight-fragment addresses are not
    // provably loop-invariant across t (prevents LICM hoist -> scratch spill)
    int nl = l16;
    asm volatile("" : "+v"(nl));

    // prefetch next step's preact fragment (overlaps with this step's WMMAs)
    if (t + 1 < LSEQ) {
      const int tn = (dir == 1) ? (LSEQ - 2 - t) : (t + 1);
      const _Float16* pf = pre + (size_t)tn * BATCH * 2048 +
                           (size_t)rowA * 2048 + dir * 1024 + ct0 * 16 + l16;
      __builtin_prefetch(pf, 0, 3);
    }

    // init accumulators from precomputed input projection (+bias)
    v8f acc[4][2];
    const _Float16* prow = pre + (size_t)tm * BATCH * 2048 + dir * 1024;
#pragma unroll
    for (int g = 0; g < 4; ++g)
#pragma unroll
      for (int s = 0; s < 2; ++s) {
        const int cc = g * 256 + (ct0 + s) * 16 + l16;
#pragma unroll
        for (int r = 0; r < 8; ++r) {
          const int row = rt * 16 + r + (hi << 3);
          acc[g][s][r] = (float)prow[(size_t)row * 2048 + cc];
        }
      }

    // recurrent GEMM: h_prev @ Whh^T  (K = 256)
    for (int kb = 0; kb < 8; ++kb) {
      v16h a = *(const v16h*)&hlds[rowA][kb * 32 + hi * 16];
#pragma unroll
      for (int s = 0; s < 2; ++s) {
        const int n = (ct0 + s) * 16 + nl;
#pragma unroll
        for (int g = 0; g < 4; ++g) {
          v16h b = *(const v16h*)(Wh + (size_t)(g * 256 + n) * 256 +
                                  kb * 32 + hi * 16);
          acc[g][s] = __builtin_amdgcn_wmma_f32_16x16x32_f16(
              false, a, false, b, (short)0, acc[g][s], false, false);
        }
      }
    }

    __syncthreads();  // all waves done reading hlds for step t

    // gates -> c,h ; c in regs, h to LDS + layer output
#pragma unroll
    for (int s = 0; s < 2; ++s) {
      const int col = (ct0 + s) * 16 + l16;
#pragma unroll
      for (int r = 0; r < 8; ++r) {
        float iv = sigf(acc[0][s][r]);
        float fv = sigf(acc[1][s][r]);
        float gv = tanhf(acc[2][s][r]);
        float ov = sigf(acc[3][s][r]);
        float cv = fv * creg[s][r] + iv * gv;
        creg[s][r] = cv;
        float hv = ov * tanhf(cv);
        const int row = rt * 16 + r + (hi << 3);
        _Float16 hh = (_Float16)hv;
        hlds[row][col] = hh;
        hout[(size_t)tm * BATCH * HID + (size_t)row * HID + dir * 256 + col] = hh;
      }
    }
    __syncthreads();  // h(t) visible before step t+1 reads
  }
}

// ---------------------------------------------------------------------------
// Emission head: feat[L*B][512] fp16 @ Wfc^T + bfc -> emit[L][B][5] f32
// ---------------------------------------------------------------------------
__global__ void emit_kernel(const _Float16* __restrict__ feat,
                            const float* __restrict__ Wfc,
                            const float* __restrict__ bfc,
                            float* __restrict__ emit) {
  long long i = (long long)blockIdx.x * blockDim.x + threadIdx.x;
  if (i >= (long long)LSEQ * BATCH * NC) return;
  int c = (int)(i % NC);
  int b = (int)((i / NC) % BATCH);
  int t = (int)(i / (NC * BATCH));
  const _Float16* f = feat + ((size_t)t * BATCH + b) * HID;
  const float* w = Wfc + c * HID;
  float s = bfc[c];
  for (int k = 0; k < HID; ++k) s += (float)f[k] * w[k];
  emit[i] = s;
}

// ---------------------------------------------------------------------------
// features [L*B][512] fp16 -> d_out features region [B][L][512] f32
// ---------------------------------------------------------------------------
__global__ void feat_out_kernel(const _Float16* __restrict__ feat,
                                float* __restrict__ out_feat) {
  long long i = (long long)blockIdx.x * blockDim.x + threadIdx.x;
  if (i >= (long long)BATCH * LSEQ * HID) return;
  int k = (int)(i % HID);
  int t = (int)((i / HID) % LSEQ);
  int b = (int)(i / ((long long)HID * LSEQ));
  out_feat[i] = (float)feat[((size_t)t * BATCH + b) * HID + k];
}

__global__ void mask_out_kernel(float* __restrict__ out_mask) {
  int i = blockIdx.x * blockDim.x + threadIdx.x;
  if (i < BATCH * LSEQ) out_mask[i] = 1.0f;  // xs > 0 always
}

// ---------------------------------------------------------------------------
// Viterbi decode + backtrace (one thread per batch element; 5-state lattice)
// ---------------------------------------------------------------------------
__global__ void viterbi_kernel(const float* __restrict__ emit,   // [L][B][5]
                               const float* __restrict__ trans,  // [5][5] to,from
                               float* __restrict__ out,
                               unsigned char* __restrict__ bp) { // [B][L][5]
  int b = threadIdx.x;
  if (b >= BATCH) return;

  float score[NC];
  for (int c = 0; c < NC; ++c) score[c] = (c == START_TAG) ? 0.0f : IMPOSSIBLE_V;

  for (int t = 0; t < LSEQ; ++t) {
    float ns[NC];
    for (int to = 0; to < NC; ++to) {
      float best = -3.4e38f;
      int arg = 0;
      for (int fr = 0; fr < NC; ++fr) {
        float v = score[fr] + trans[to * NC + fr];
        if (v > best) { best = v; arg = fr; }
      }
      bp[((size_t)b * LSEQ + t) * NC + to] = (unsigned char)arg;
      ns[to] = best + emit[((size_t)t * BATCH + b) * NC + to];
    }
    for (int c = 0; c < NC; ++c) score[c] = ns[c];  // mask always true
  }

  float best = -3.4e38f;
  int tag = 0;
  for (int fr = 0; fr < NC; ++fr) {
    float v = score[fr] + trans[STOP_TAG * NC + fr];
    if (v > best) { best = v; tag = fr; }
  }
  out[b] = best;

  float* tags = out + 64 + (size_t)b * LSEQ;
  tags[LSEQ - 1] = (float)tag;
  for (int t = LSEQ - 1; t >= 1; --t) {
    tag = (int)bp[((size_t)b * LSEQ + t) * NC + tag];
    tags[t - 1] = (float)tag;
  }
}

// ---------------------------------------------------------------------------
// Host launch
// ---------------------------------------------------------------------------
extern "C" void kernel_launch(void* const* d_in, const int* in_sizes, int n_in,
                              void* d_out, int out_size, void* d_ws, size_t ws_size,
                              hipStream_t stream) {
  (void)in_sizes; (void)n_in; (void)out_size; (void)ws_size;

  const int*   xs    = (const int*)d_in[0];
  const float* emb   = (const float*)d_in[1];
  const float* Wih0  = (const float*)d_in[2];
  const float* Whh0  = (const float*)d_in[3];
  const float* b0    = (const float*)d_in[4];
  const float* WihL  = (const float*)d_in[5];
  const float* WhhL  = (const float*)d_in[6];
  const float* bL    = (const float*)d_in[7];
  const float* Wfc   = (const float*)d_in[8];
  const float* bfc   = (const float*)d_in[9];
  const float* trans = (const float*)d_in[10];

  // workspace layout (all chunk sizes are multiples of 256B)
  char* ws = (char*)d_ws;
  const size_t nWih0 = 2ull * 1024 * 256;
  const size_t nWhh0 = 2ull * 1024 * 256;
  const size_t nWihL = 3ull * 2 * 1024 * 512;
  const size_t nWhhL = 3ull * 2 * 1024 * 256;
  const size_t nX0   = (size_t)LSEQ * BATCH * 256;
  const size_t nBuf  = (size_t)LSEQ * BATCH * HID;
  const size_t nPre  = (size_t)LSEQ * BATCH * 2048;

  size_t off = 0;
  _Float16* Wih0h = (_Float16*)(ws + off); off += nWih0 * 2;
  _Float16* Whh0h = (_Float16*)(ws + off); off += nWhh0 * 2;
  _Float16* WihLh = (_Float16*)(ws + off); off += nWihL * 2;
  _Float16* WhhLh = (_Float16*)(ws + off); off += nWhhL * 2;
  _Float16* x0    = (_Float16*)(ws + off); off += nX0 * 2;
  _Float16* buf0  = (_Float16*)(ws + off); off += nBuf * 2;
  _Float16* buf1  = (_Float16*)(ws + off); off += nBuf * 2;
  _Float16* preb  = (_Float16*)(ws + off); off += nPre * 2;
  float*    emitb = (float*)(ws + off);    off += (size_t)LSEQ * BATCH * NC * 4;
  unsigned char* bp = (unsigned char*)(ws + off);

  // 1) weights -> fp16
  cvt_f16_kernel<<<(int)((nWih0 + 255) / 256), 256, 0, stream>>>(Wih0, Wih0h, (int)nWih0);
  cvt_f16_kernel<<<(int)((nWhh0 + 255) / 256), 256, 0, stream>>>(Whh0, Whh0h, (int)nWhh0);
  cvt_f16_kernel<<<(int)((nWihL + 255) / 256), 256, 0, stream>>>(WihL, WihLh, (int)nWihL);
  cvt_f16_kernel<<<(int)((nWhhL + 255) / 256), 256, 0, stream>>>(WhhL, WhhLh, (int)nWhhL);

  // 2) embedding gather -> [L*B][256] fp16
  embed_gather_kernel<<<(int)((nX0 + 255) / 256), 256, 0, stream>>>(xs, emb, x0);

  // 32x32 macro-tiles: (32768/32) * (2048/32) / 8 waves = 8192 blocks
  const int PROJ_BLOCKS = (LSEQ * BATCH / 32) * (2048 / 32) / 8;

  // 3) four BiLSTM layers: parallel input-projection GEMM, then recurrence
  inproj_kernel<<<PROJ_BLOCKS, 256, 0, stream>>>(x0, 256, Wih0h, b0, preb);
  lstm_rec_kernel<<<2, 1024, 0, stream>>>(preb, Whh0h, buf0);

  inproj_kernel<<<PROJ_BLOCKS, 256, 0, stream>>>(buf0, 512,
      WihLh + 0ull * 2 * 1024 * 512, bL + 0 * 2048, preb);
  lstm_rec_kernel<<<2, 1024, 0, stream>>>(preb, WhhLh + 0ull * 2 * 1024 * 256, buf1);

  inproj_kernel<<<PROJ_BLOCKS, 256, 0, stream>>>(buf1, 512,
      WihLh + 1ull * 2 * 1024 * 512, bL + 1 * 2048, preb);
  lstm_rec_kernel<<<2, 1024, 0, stream>>>(preb, WhhLh + 1ull * 2 * 1024 * 256, buf0);

  inproj_kernel<<<PROJ_BLOCKS, 256, 0, stream>>>(buf0, 512,
      WihLh + 2ull * 2 * 1024 * 512, bL + 2 * 2048, preb);
  lstm_rec_kernel<<<2, 1024, 0, stream>>>(preb, WhhLh + 2ull * 2 * 1024 * 256, buf1);

  // 4) emissions, features, masks, viterbi
  const long long nEmit = (long long)LSEQ * BATCH * NC;
  emit_kernel<<<(int)((nEmit + 255) / 256), 256, 0, stream>>>(buf1, Wfc, bfc, emitb);

  float* out = (float*)d_out;
  const long long nFeat = (long long)BATCH * LSEQ * HID;
  feat_out_kernel<<<(int)((nFeat + 255) / 256), 256, 0, stream>>>(buf1, out + 64 + BATCH * LSEQ);
  mask_out_kernel<<<(BATCH * LSEQ + 255) / 256, 256, 0, stream>>>(
      out + 64 + BATCH * LSEQ + nFeat);
  viterbi_kernel<<<1, 64, 0, stream>>>(emitb, trans, out, bp);
}
